// LossEvaluator_79431125172241
// MI455X (gfx1250) — compile-verified, matrix operand order checked by reading, not run on hardware
//
#include <hip/hip_runtime.h>
#include <hip/hip_bf16.h>
#include <stdint.h>

typedef _Float16 f16;
typedef __attribute__((ext_vector_type(16))) _Float16 v16h;
typedef __attribute__((ext_vector_type(8)))  _Float16 v8h;
typedef __attribute__((ext_vector_type(8)))  float    v8f;

#define B_    8
#define S_    128
#define NS_   64
#define NC_   8
#define MT_   8
#define ZD_   64
#define DC_   128
#define H_    256
#define M_    65536          /* B*S*NS */
#define KPAD1 352            /* pad(257 + 64) to 11*32 */
#define LOG2PI_F 1.8378770664093453f

/* ---------------- workspace byte offsets (lifetime-based reuse) ------------- */
static const size_t OFF_X    = 0;            /* f16 M x 352 (dead after h GEMM) */
static const size_t OFF_E1   = 0;            /* f16 M x 256 (reuses X)          */
static const size_t OFF_EMIT = 33554432ull;  /* f32 M x 16  (inside X region)   */
static const size_t OFF_Z0H  = 46137344ull;  /* f16 M x 64                      */
static const size_t OFF_H    = 54525952ull;  /* f16 M x 256 (dead after gp)     */
static const size_t OFF_ZLOC = 54525952ull;  /* f32 M x 64  (reuses H)          */
static const size_t OFF_RP   = 71303168ull;  /* f16 M x 64                      */
static const size_t OFF_GP   = 88080384ull;  /* f32 M x 128 (dead after K3)     */
static const size_t OFF_SPRE = 88080384ull;  /* f32 M x 64  (reuses GP)         */
static const size_t OFF_ZSH  = 104857600ull; /* f16 M x 64                      */
static const size_t OFF_ZLP  = 113246208ull; /* f32 M                           */
static const size_t OFF_ZLIN = 121634816ull; /* f32 M x 64 (dead after K3)      */
static const size_t OFF_LOGP = 121634816ull; /* f32 M (reuses ZLIN)             */
static const size_t OFF_COST = 121896960ull; /* f32 M                           */
static const size_t OFF_PART = 122159104ull; /* f32 B*S                         */
static const size_t OFF_W    = 138412032ull; /* packed f16 weights              */
static const size_t OFF_WCAT = OFF_W + 0;
static const size_t OFF_WGP  = OFF_W + 180224;
static const size_t OFF_WZ   = OFF_W + 245760;
static const size_t OFF_WSP  = OFF_W + 253952;
static const size_t OFF_WE1  = OFF_W + 262144;
static const size_t OFF_WE2  = OFF_W + 294912;
static const size_t OFF_BCAT = OFF_W + 303104; /* f32 128 */

/* ---------------- small device helpers ------------------------------------- */
__device__ __forceinline__ unsigned pcg_hash(unsigned v) {
  unsigned s = v * 747796405u + 2891336453u;
  unsigned w = ((s >> ((s >> 28) + 4u)) ^ s) * 277803737u;
  return (w >> 22) ^ w;
}
__device__ __forceinline__ float u01f(unsigned h) {
  return ((float)(h >> 8) + 0.5f) * (1.0f / 16777216.0f);
}
__device__ __forceinline__ float gaussian_rng(unsigned idx) {
  float a = u01f(pcg_hash(idx ^ 0x68E31DA4u));
  float b = u01f(pcg_hash(idx ^ 0xB5297A4Du));
  return sqrtf(-2.0f * logf(a)) * cosf(6.28318530718f * b);
}
__device__ __forceinline__ float softplusf(float x) {
  return x > 20.f ? x : log1pf(expf(x));
}
__device__ __forceinline__ float sigmoidf(float x) {
  return 1.f / (1.f + expf(-x));
}

/* ---------------- weight packing into WMMA B-fragment layout ----------------
 * Packed layout: dst[((kb*N + n)*32) + half*16 + e], where the K index held by
 * lane-half `half`, element e is  k = kb*32 + half*8 + (e<8 ? e : e+8)
 * (matches 16-bit A/B fragment layout in CDNA5 ISA 7.12.2).       */
__global__ __launch_bounds__(256) void pack_kcat_kernel(
    const float* __restrict__ srcA, const float* __restrict__ srcB,
    f16* __restrict__ dst, int KA, int KB2, int Kpad, int N) {
  int idx = blockIdx.x * 256 + threadIdx.x;
  int total = (Kpad >> 5) * N * 32;
  if (idx >= total) return;
  int j = idx & 31;
  int rem = idx >> 5;
  int n = rem % N;
  int kb = rem / N;
  int h = j >> 4, e = j & 15;
  int k = kb * 32 + h * 8 + (e < 8 ? e : e + 8);
  float v = 0.f;
  if (k < KA) v = srcA[(size_t)k * N + n];
  else if (srcB && k < KA + KB2) v = srcB[(size_t)(k - KA) * N + n];
  dst[idx] = (f16)v;
}

__global__ __launch_bounds__(256) void pack_ncat_kernel(
    const float* __restrict__ srcA, const float* __restrict__ srcB,
    f16* __restrict__ dst, int K, int NA, int N) {
  int idx = blockIdx.x * 256 + threadIdx.x;
  int total = (K >> 5) * N * 32;
  if (idx >= total) return;
  int j = idx & 31;
  int rem = idx >> 5;
  int n = rem % N;
  int kb = rem / N;
  int h = j >> 4, e = j & 15;
  int k = kb * 32 + h * 8 + (e < 8 ? e : e + 8);
  float v = (n < NA) ? srcA[(size_t)k * NA + n]
                     : srcB[(size_t)k * (N - NA) + (n - NA)];
  dst[idx] = (f16)v;
}

__global__ __launch_bounds__(128) void biascat_kernel(
    const float* __restrict__ a, const float* __restrict__ b, float* __restrict__ dst) {
  int i = threadIdx.x;
  dst[i] = (i < 64) ? a[i] : b[i - 64];
}

/* ---------------- feature build: Xcat f16 [M x 352], Z0 f16 [M x 64] -------- */
__global__ __launch_bounds__(64) void build_x_kernel(
    const float* __restrict__ conv, const float* __restrict__ z0,
    const float* __restrict__ rates, const float* __restrict__ acr_in,
    const float* __restrict__ tmarg, const float* __restrict__ ots,
    const float* __restrict__ otr, f16* __restrict__ X, f16* __restrict__ Z0h) {
  int r = blockIdx.x;
  int tid = threadIdx.x;
  int bs = r / NS_;
  int b = bs / S_;
  __shared__ float red[64];

  int c = tid >> 3, t = tid & 7;
  float acr_c = acr_in[bs * NC_ + c];
  float sz = ots[((size_t)r * NC_ + c) * MT_ + t];
  float margin = sz / acr_c;                    /* LEV == 1.0 */
  red[tid] = fabsf(margin);
  __syncthreads();
  for (int off = 32; off > 0; off >>= 1) {
    if (tid < off) red[tid] += red[tid + off];
    __syncthreads();
  }
  float tm = tmarg[r];
  float inv_tm = 1.0f / tm;

  f16* xr = X + (size_t)r * KPAD1;
  xr[tid]       = (f16)conv[(size_t)bs * DC_ + tid];
  xr[64 + tid]  = (f16)conv[(size_t)bs * DC_ + 64 + tid];
  xr[128 + tid] = (f16)(margin * inv_tm);
  if (tid == 0) xr[192] = (f16)((tm - red[0]) * inv_tm);
  size_t rbase = ((size_t)(b * S_ + (S_ - 1)) * NC_ + c) * 2;
  float mr = 0.5f * (rates[rbase] + rates[rbase + 1]);
  xr[193 + tid] = (f16)(otr[((size_t)r * NC_ + c) * MT_ + t] / mr);
  float zv = z0[(size_t)r * ZD_ + tid];
  xr[257 + tid] = (f16)zv;
  Z0h[(size_t)r * ZD_ + tid] = (f16)zv;
  if (tid < 31) xr[321 + tid] = (f16)0.f;       /* zero pad 321..351 */
}

/* ---------------- generic WMMA GEMM: one wave = 16 x (CT*16) strip ----------
 * A: f16 row-major [M x Kpad]; Bp: packed fragments; bias f32[N].
 * Loads the A fragment ONCE per k-step and issues CT WMMAs against CT B
 * fragments -> 4x less A traffic, 4x higher wmma density vs 1 tile/wave.    */
template <int CT>
__global__ __launch_bounds__(32) void wmma_gemm_kernel(
    const f16* __restrict__ A, const f16* __restrict__ Bp,
    const float* __restrict__ bias, float* __restrict__ Cf32,
    f16* __restrict__ Cf16, int N, int Kpad, int relu) {
  int lane = threadIdx.x;
  int halfSel = lane >> 4;
  int mrow = lane & 15;
  int rt = blockIdx.x, ctg = blockIdx.y;
  int KB = Kpad >> 5;

  const f16* arow = A + (size_t)(rt * 16 + mrow) * Kpad + halfSel * 8;
  /* B fragment base for first tile of this strip; tile stride = 16*32 elems */
  const f16* bbase = Bp + ((size_t)(ctg * CT * 16 + mrow)) * 32 + halfSel * 16;
  size_t bstep = (size_t)N * 32;

  v8f acc[CT];
#pragma unroll
  for (int tt = 0; tt < CT; ++tt) acc[tt] = (v8f){};

  for (int kb = 0; kb < KB; ++kb) {
    v8h alo = *(const v8h*)(arow);
    v8h ahi = *(const v8h*)(arow + 16);
    v16h a = __builtin_shufflevector(alo, ahi, 0, 1, 2, 3, 4, 5, 6, 7,
                                     8, 9, 10, 11, 12, 13, 14, 15);
#pragma unroll
    for (int tt = 0; tt < CT; ++tt) {
      const f16* bp = bbase + (size_t)tt * 512;
      v8h blo = *(const v8h*)(bp);
      v8h bhi = *(const v8h*)(bp + 8);
      v16h bfr = __builtin_shufflevector(blo, bhi, 0, 1, 2, 3, 4, 5, 6, 7,
                                         8, 9, 10, 11, 12, 13, 14, 15);
      acc[tt] = __builtin_amdgcn_wmma_f32_16x16x32_f16(false, a, false, bfr,
                                                       (short)0, acc[tt],
                                                       false, false);
    }
    arow += 32;
    bbase += bstep;
  }
#pragma unroll
  for (int tt = 0; tt < CT; ++tt) {
    int n = ctg * CT * 16 + tt * 16 + mrow;
    float bv = bias ? bias[n] : 0.0f;
#pragma unroll
    for (int v = 0; v < 8; ++v) {
      int m = rt * 16 + halfSel * 8 + v; /* D layout: lanes16-31 hold M=v+8 */
      float val = acc[tt][v] + bv;
      if (relu) val = val > 0.f ? val : 0.f;
      size_t o = (size_t)m * N + n;
      if (Cf32) Cf32[o] = val;
      if (Cf16) Cf16[o] = (f16)val;
    }
  }
}

/* ---------------- gate / z_loc / relu(prop) -------------------------------- */
__global__ __launch_bounds__(256) void gate_zloc_kernel(
    const float* __restrict__ gp, const float* __restrict__ zlin,
    float* __restrict__ zloc, f16* __restrict__ rp) {
  size_t i = (size_t)blockIdx.x * 256 + threadIdx.x;
  size_t rr = i >> 6;
  int d = (int)(i & 63);
  float g = sigmoidf(gp[rr * 128 + d]);
  float p = gp[rr * 128 + 64 + d];
  float zl = zlin[i];
  zloc[i] = (1.f - g) * zl + g * p;
  rp[i] = (f16)(p > 0.f ? p : 0.f);
}

/* ---------------- sampling + per-row z logprob ------------------------------ */
__global__ __launch_bounds__(64) void sample_kernel(
    const float* __restrict__ spre, const float* __restrict__ zloc,
    float* __restrict__ zs_out, f16* __restrict__ zs_h, float* __restrict__ zlogp) {
  int r = blockIdx.x;
  int d = threadIdx.x;
  size_t i = (size_t)r * ZD_ + d;
  float sc = softplusf(spre[i]) + 1e-4f;
  float eps = gaussian_rng(0x1234567u + (unsigned)i);
  float zs = zloc[i] + sc * eps;
  zs_out[i] = zs;
  zs_h[i] = (f16)zs;
  __shared__ float red[64];
  red[d] = -0.5f * eps * eps - logf(sc) - 0.5f * LOG2PI_F;
  __syncthreads();
  for (int off = 32; off > 0; off >>= 1) {
    if (d < off) red[d] += red[d + off];
    __syncthreads();
  }
  if (d == 0) zlogp[r] = red[0];
}

/* ---------------- per-row trading simulation ------------------------------- */
__global__ __launch_bounds__(256) void sim_kernel(
    const float* __restrict__ rates, const float* __restrict__ acr_in,
    const float* __restrict__ tmarg, const float* __restrict__ ots_in,
    const float* __restrict__ otr_in, const float* __restrict__ emit,
    const float* __restrict__ zlogp, float* __restrict__ tm_out,
    float* __restrict__ ots_out, float* __restrict__ otr_out,
    float* __restrict__ logprobs, float* __restrict__ costs) {
  int r = blockIdx.x * 256 + threadIdx.x;
  if (r >= M_) return;
  int bs = r / NS_;
  float tm = tmarg[r];

  float acr[NC_], r0[NC_], r1[NC_], firstv[NC_], lastv[NC_], closer[NC_];
  float psize[NC_], used_m[NC_];
  float sum_used_m = 0.f, total_used = 0.f, av = tm;

#pragma unroll
  for (int c = 0; c < NC_; ++c) {
    acr[c] = acr_in[bs * NC_ + c];
    r0[c] = rates[((size_t)bs * NC_ + c) * 2];
    r1[c] = rates[((size_t)bs * NC_ + c) * 2 + 1];
    size_t base = ((size_t)r * NC_ + c) * MT_;
    firstv[c] = ots_in[base];
    lastv[c] = ots_in[base + MT_ - 1];
    closer[c] = firstv[c] > 0.f ? r0[c] : (firstv[c] < 0.f ? r1[c] : 1.f);
    float ps = 0.f, us = 0.f;
#pragma unroll
    for (int t = 0; t < MT_; ++t) {
      float m = ots_in[base + t] / acr[c];          /* /LEV, LEV==1 */
      ps += m;
      us += fabsf(m);
      av += (ots_in[base + t] / acr[c]) * (1.f - closer[c] / otr_in[base + t]);
    }
    psize[c] = ps;
    used_m[c] = fabsf(ps) / acr[c];                 /* /(LEV*acr)   */
    sum_used_m += used_m[c];
    total_used += us;
  }
  bool closeout = av < 0.5f * total_used;

  float elp = 0.f;
  for (int c = 0; c < NC_; ++c) {
    float el = emit[(size_t)r * 16 + 2 * c];
    float fr = tanhf(emit[(size_t)r * 16 + 2 * c + 1]);
    float u = u01f(pcg_hash(0x85EBCA6Bu ^ (unsigned)(r * NC_ + c)));
    float es_samp = (u < sigmoidf(el)) ? 1.f : 0.f;
    float elogp = es_samp * el - softplusf(el);
    bool opposite = fr * firstv[c] <= 0.f;
    bool emask = es_samp == 1.f;
    bool ignore_add = (!opposite) && (lastv[c] != 0.f);
    bool add_tr = (!opposite) && (lastv[c] == 0.f);
    float openr = fr > 0.f ? r1[c] : (fr < 0.f ? r0[c] : 1.f);

    float tun = tm - sum_used_m;
    float availv = tun + (opposite ? used_m[c] : 0.f);
    float es = fr * availv * acr[c];
    if (!emask) es = 0.f;
    if (closeout) es = -psize[c];

    float cs[MT_], cr[MT_];
    size_t base = ((size_t)r * NC_ + c) * MT_;
#pragma unroll
    for (int t = 0; t < MT_; ++t) { cs[t] = ots_in[base + t]; cr[t] = otr_in[base + t]; }
    if (add_tr) {
#pragma unroll
      for (int t = MT_ - 1; t > 0; --t) { cs[t] = cs[t - 1]; cr[t] = cr[t - 1]; }
      cs[0] = es;
      cr[0] = openr;
    }
    float left = opposite ? es : 0.f;
    bool covered = false;
#pragma unroll
    for (int j = MT_ - 1; j >= 0; --j) {
      float diffs = left + cs[j];
      bool cov_now = (!covered) && (diffs * left <= 0.f);
      covered = covered || cov_now;
      float closed = fabsf(cov_now ? left : cs[j]);
      tm += closed / acr[c] * (1.f - closer[c] / cr[j]);
      cs[j] = cov_now ? diffs : 0.f;
      cr[j] = cov_now ? cr[j] : 0.f;
      left = cov_now ? 0.f : diffs;
    }
    if (!covered) { cs[0] = left; cr[0] = openr; }
#pragma unroll
    for (int t = 0; t < MT_; ++t) { ots_out[base + t] = cs[t]; otr_out[base + t] = cr[t]; }
    if (!(closeout || ignore_add)) elp += elogp;
  }
  tm_out[r] = tm;
  float cost = -logf(tm);
  costs[r] = cost;
  logprobs[r] = zlogp[r] + elp;
}

/* ---------------- loss reduction ------------------------------------------- */
__global__ __launch_bounds__(64) void loss_bs_kernel(
    const float* __restrict__ costs, const float* __restrict__ logprobs,
    float* __restrict__ partial) {
  int bsi = blockIdx.x;
  int ns = threadIdx.x;
  size_t i = (size_t)bsi * NS_ + ns;
  __shared__ float red[64];
  float cost = costs[i];
  red[ns] = cost;
  __syncthreads();
  for (int off = 32; off > 0; off >>= 1) {
    if (ns < off) red[ns] += red[ns + off];
    __syncthreads();
  }
  float base = red[0] * (1.0f / NS_);
  __syncthreads();
  red[ns] = logprobs[i] * (cost - base) + cost;
  __syncthreads();
  for (int off = 32; off > 0; off >>= 1) {
    if (ns < off) red[ns] += red[ns + off];
    __syncthreads();
  }
  if (ns == 0) partial[bsi] = red[0] * (1.0f / NS_);
}

__global__ __launch_bounds__(256) void loss_final_kernel(
    const float* __restrict__ partial, float* __restrict__ out) {
  __shared__ float red[256];
  float s = 0.f;
  for (int i = threadIdx.x; i < B_ * S_; i += 256) s += partial[i];
  red[threadIdx.x] = s;
  __syncthreads();
  for (int off = 128; off > 0; off >>= 1) {
    if (threadIdx.x < off) red[threadIdx.x] += red[threadIdx.x + off];
    __syncthreads();
  }
  if (threadIdx.x == 0) out[0] = red[0];
}

/* ---------------- host launcher -------------------------------------------- */
extern "C" void kernel_launch(void* const* d_in, const int* in_sizes, int n_in,
                              void* d_out, int out_size, void* d_ws, size_t ws_size,
                              hipStream_t stream) {
  (void)in_sizes; (void)n_in; (void)out_size; (void)ws_size;
  const float* conv = (const float*)d_in[0];
  const float* z0   = (const float*)d_in[1];
  const float* rates = (const float*)d_in[2];
  const float* acr  = (const float*)d_in[3];
  const float* tmarg = (const float*)d_in[4];
  const float* ots  = (const float*)d_in[5];
  const float* otr  = (const float*)d_in[6];
  const float* W1 = (const float*)d_in[7];
  const float* U1 = (const float*)d_in[8];
  const float* b1 = (const float*)d_in[9];
  const float* Wg = (const float*)d_in[10];
  const float* bg = (const float*)d_in[11];
  const float* Wp = (const float*)d_in[12];
  const float* bp = (const float*)d_in[13];
  const float* Wz = (const float*)d_in[14];
  const float* bz = (const float*)d_in[15];
  const float* Wsw = (const float*)d_in[16];
  const float* bsb = (const float*)d_in[17];
  const float* We1 = (const float*)d_in[18];
  const float* be1 = (const float*)d_in[19];
  const float* We2 = (const float*)d_in[20];
  const float* be2 = (const float*)d_in[21];

  char* ws = (char*)d_ws;
  f16*  X    = (f16*)(ws + OFF_X);
  f16*  Z0h  = (f16*)(ws + OFF_Z0H);
  f16*  Hh   = (f16*)(ws + OFF_H);
  float* GP  = (float*)(ws + OFF_GP);
  float* ZLIN = (float*)(ws + OFF_ZLIN);
  float* ZLOC = (float*)(ws + OFF_ZLOC);
  f16*  RP   = (f16*)(ws + OFF_RP);
  float* SPRE = (float*)(ws + OFF_SPRE);
  f16*  ZSH  = (f16*)(ws + OFF_ZSH);
  float* ZLP = (float*)(ws + OFF_ZLP);
  f16*  E1   = (f16*)(ws + OFF_E1);
  float* EMIT = (float*)(ws + OFF_EMIT);
  float* LOGP = (float*)(ws + OFF_LOGP);
  float* COST = (float*)(ws + OFF_COST);
  float* PART = (float*)(ws + OFF_PART);
  f16* WCAT = (f16*)(ws + OFF_WCAT);
  f16* WGPp = (f16*)(ws + OFF_WGP);
  f16* WZp  = (f16*)(ws + OFF_WZ);
  f16* WSp  = (f16*)(ws + OFF_WSP);
  f16* WE1p = (f16*)(ws + OFF_WE1);
  f16* WE2p = (f16*)(ws + OFF_WE2);
  float* BCAT = (float*)(ws + OFF_BCAT);

  float* out = (float*)d_out;
  float* out_zs  = out + 1;
  float* out_tm  = out + 1 + (size_t)M_ * ZD_;
  float* out_ots = out_tm + M_;
  float* out_otr = out_ots + (size_t)M_ * NC_ * MT_;

  /* 1) pack all weights into WMMA B-fragment layout (f16) */
  pack_kcat_kernel<<<(11 * 256 * 32 + 255) / 256, 256, 0, stream>>>(W1, U1, WCAT, 257, 64, KPAD1, 256);
  pack_ncat_kernel<<<(8 * 128 * 32 + 255) / 256, 256, 0, stream>>>(Wg, Wp, WGPp, 256, 64, 128);
  pack_kcat_kernel<<<(2 * 64 * 32 + 255) / 256, 256, 0, stream>>>(Wz, nullptr, WZp, 64, 0, 64, 64);
  pack_kcat_kernel<<<(2 * 64 * 32 + 255) / 256, 256, 0, stream>>>(Wsw, nullptr, WSp, 64, 0, 64, 64);
  pack_kcat_kernel<<<(2 * 256 * 32 + 255) / 256, 256, 0, stream>>>(We1, nullptr, WE1p, 64, 0, 64, 256);
  pack_kcat_kernel<<<(8 * 16 * 32 + 255) / 256, 256, 0, stream>>>(We2, nullptr, WE2p, 256, 0, 256, 16);
  biascat_kernel<<<1, 128, 0, stream>>>(bg, bp, BCAT);

  /* 2) build fused input features (conv|margins|tun|rel_rates|z0, f16) */
  build_x_kernel<<<M_, 64, 0, stream>>>(conv, z0, rates, acr, tmarg, ots, otr, X, Z0h);

  /* 3) GEMM chain on WMMA: each wave owns a 16x64 strip (CT=4) */
  wmma_gemm_kernel<4><<<dim3(M_ / 16, 256 / 64), 32, 0, stream>>>(X, WCAT, b1, nullptr, Hh, 256, KPAD1, 1);
  wmma_gemm_kernel<4><<<dim3(M_ / 16, 128 / 64), 32, 0, stream>>>(Hh, WGPp, BCAT, GP, nullptr, 128, 256, 0);
  wmma_gemm_kernel<4><<<dim3(M_ / 16, 64 / 64), 32, 0, stream>>>(Z0h, WZp, bz, ZLIN, nullptr, 64, 64, 0);
  gate_zloc_kernel<<<(M_ * 64) / 256, 256, 0, stream>>>(GP, ZLIN, ZLOC, RP);
  wmma_gemm_kernel<4><<<dim3(M_ / 16, 64 / 64), 32, 0, stream>>>(RP, WSp, bsb, SPRE, nullptr, 64, 64, 0);
  sample_kernel<<<M_, 64, 0, stream>>>(SPRE, ZLOC, out_zs, ZSH, ZLP);
  wmma_gemm_kernel<4><<<dim3(M_ / 16, 256 / 64), 32, 0, stream>>>(ZSH, WE1p, be1, nullptr, E1, 256, 64, 1);
  wmma_gemm_kernel<1><<<dim3(M_ / 16, 1), 32, 0, stream>>>(E1, WE2p, be2, EMIT, nullptr, 16, 256, 0);

  /* 4) per-row trading simulation + outputs */
  sim_kernel<<<M_ / 256, 256, 0, stream>>>(rates, acr, tmarg, ots, otr, EMIT, ZLP,
                                           out_tm, out_ots, out_otr, LOGP, COST);

  /* 5) loss reduction */
  loss_bs_kernel<<<B_ * S_, 64, 0, stream>>>(COST, LOGP, PART);
  loss_final_kernel<<<1, 256, 0, stream>>>(PART, out);
}
